// GemmaAttention_29480655519827
// MI455X (gfx1250) — matrix operand order repaired and checked by reading, not compiled, not run on hardware
//
#include <hip/hip_runtime.h>
#include <hip/hip_bf16.h>

typedef __attribute__((ext_vector_type(16))) __bf16 v16bf;
typedef __attribute__((ext_vector_type(8)))  float  v8f;

// ---------------------------------------------------------------------------
// CDNA5 async global->LDS copy (bypasses VGPRs, tracked on ASYNCcnt).
// Generic LDS pointer low 32 bits == wave-relative LDS byte address (AS3).
// ---------------------------------------------------------------------------
__device__ __forceinline__ void cp_async_b128(void* lds, const void* g) {
  const unsigned l = (unsigned)(unsigned long long)lds;
  asm volatile("global_load_async_to_lds_b128 %0, %1, off"
               :: "v"(l), "v"(g) : "memory");
}

template <int N>
__device__ __forceinline__ void wait_asynccnt() {
  asm volatile("s_wait_asynccnt %0" :: "n"(N) : "memory");
}

// ---------------------------------------------------------------------------
// Generic bf16-WMMA GEMM:  C[m][n] = alpha * sum_k A[m][k]*B(k,n)  + add[n]
//   BT=true : B stored row-major as Bt[n][k]   (weights, K^T)
//   BT=false: B stored row-major as B[k][n]    (attn @ V)
// Tiles staged as f32 via GLOBAL_LOAD_ASYNC_TO_LDS_B128, double-buffered;
// f32->bf16 conversion happens at fragment gather.
// Batched over blockIdx.z:  off = (z/headDiv)*s1 + (z%headDiv)*s2
// Requires: M%128==0, N%128==0, K%32==0 (true for all launches below).
// ---------------------------------------------------------------------------
template<bool BT>
__global__ __launch_bounds__(256)
void gemm_wmma(const float* __restrict__ A, int lda, long sA1, long sA2,
               const float* __restrict__ Bm, int ldb, long sB1, long sB2,
               float* __restrict__ C, int ldc, long sC1, long sC2,
               int M, int N, int K,
               const float* __restrict__ nAdd, int addDiv, long addStride,
               float alpha, int headDiv)
{
  // A tile: [128 rows][32 k] f32, row stride 36 (144B: 16B-aligned, 36-bank)
  // B tile: BT -> [n:128][k:32] stride 36 ; BN -> [k:32][n:128] stride 132
  __shared__ float As[2][128][36];
  __shared__ float Bs[2][128 * 36];

  const int z = blockIdx.z;
  const float* Ap = A  + (long)(z / headDiv) * sA1 + (long)(z % headDiv) * sA2;
  const float* Bp = Bm + (long)(z / headDiv) * sB1 + (long)(z % headDiv) * sB2;
  float*       Cp = C  + (long)(z / headDiv) * sC1 + (long)(z % headDiv) * sC2;
  const float* addp = nAdd ? (nAdd + (long)(z / addDiv) * addStride) : nullptr;

  const int row0 = blockIdx.y * 128;
  const int col0 = blockIdx.x * 128;

  const int tid  = threadIdx.x;
  const int lane = tid & 31;
  const int wid  = tid >> 5;        // 8 waves
  const int wM   = (wid & 3) * 32;  // 4 waves along M, 32 rows each
  const int wN   = (wid >> 2) * 64; // 2 waves along N, 64 cols each
  const int lm   = lane & 15;
  const int lh   = lane >> 4;

  v8f acc[2][4] = {};

  const int c4 = (tid & 7) * 4;     // float4 column group for staging
  const int r0 = tid >> 3;          // 0..31

  // Issue one tile's async DMA: 8 x B128 per thread (4 A + 4 B).
  auto issue_tile = [&](int k0, int buf) {
#pragma unroll
    for (int it = 0; it < 4; ++it) {
      const int r = r0 + 32 * it;
      cp_async_b128(&As[buf][r][c4], Ap + (long)(row0 + r) * lda + k0 + c4);
    }
    if (BT) {
#pragma unroll
      for (int it = 0; it < 4; ++it) {
        const int r = r0 + 32 * it;
        cp_async_b128(&Bs[buf][r * 36 + c4],
                      Bp + (long)(col0 + r) * ldb + k0 + c4);
      }
    } else {
#pragma unroll
      for (int it = 0; it < 4; ++it) {
        const int n0 = c4 + it * 32;   // 0..124 step 4
        cp_async_b128(&Bs[buf][r0 * 132 + n0],
                      Bp + (long)(k0 + r0) * ldb + col0 + n0);
      }
    }
  };

  const int NT = K >> 5;
  issue_tile(0, 0);

  for (int t = 0; t < NT; ++t) {
    const int cur = t & 1;
    if (t + 1 < NT) {
      issue_tile((t + 1) << 5, cur ^ 1);
      wait_asynccnt<8>();   // tile t's 8 ops done; tile t+1 still in flight
    } else {
      wait_asynccnt<0>();
    }
    __syncthreads();        // all waves' tile-t DMA complete

    const float (*Asc)[36] = As[cur];
    const float* Bsc = Bs[cur];

    // ---- A fragments (16x32: lane m holds K=16*lh .. +15), cvt f32->bf16 ----
    v16bf af[2];
#pragma unroll
    for (int fm = 0; fm < 2; ++fm) {
      const int mr = wM + fm * 16 + lm;
#pragma unroll
      for (int j = 0; j < 8; ++j) {
        const int kb = 16 * lh + 2 * j;
        af[fm][2 * j]     = (__bf16)Asc[mr][kb];
        af[fm][2 * j + 1] = (__bf16)Asc[mr][kb + 1];
      }
    }
    // ---- B fragments (32x16: lane n holds K=16*lh .. +15) ----
    v16bf bfr[4];
#pragma unroll
    for (int fn = 0; fn < 4; ++fn) {
      const int nc = wN + fn * 16 + lm;
#pragma unroll
      for (int j = 0; j < 8; ++j) {
        const int kb = 16 * lh + 2 * j;
        if (BT) {
          bfr[fn][2 * j]     = (__bf16)Bsc[nc * 36 + kb];
          bfr[fn][2 * j + 1] = (__bf16)Bsc[nc * 36 + kb + 1];
        } else {
          bfr[fn][2 * j]     = (__bf16)Bsc[kb * 132 + nc];
          bfr[fn][2 * j + 1] = (__bf16)Bsc[(kb + 1) * 132 + nc];
        }
      }
    }
#pragma unroll
    for (int fm = 0; fm < 2; ++fm)
#pragma unroll
      for (int fn = 0; fn < 4; ++fn)
        acc[fm][fn] = __builtin_amdgcn_wmma_f32_16x16x32_bf16(
            false, af[fm], false, bfr[fn], (short)0, acc[fm][fn], false, false);

    __syncthreads();        // everyone done reading buf 'cur' before overwrite
  }

  // ---- epilogue: scale + per-column add; C/D layout m = r + 8*lh, n = lm ----
#pragma unroll
  for (int fm = 0; fm < 2; ++fm) {
#pragma unroll
    for (int fn = 0; fn < 4; ++fn) {
      const int gm0 = row0 + wM + fm * 16;
      const int gn  = col0 + wN + fn * 16 + lm;
      const float add = addp ? addp[gn] : 0.0f;
#pragma unroll
      for (int r = 0; r < 8; ++r) {
        const int gm = gm0 + r + 8 * lh;
        Cp[(long)gm * ldc + gn] = acc[fm][fn][r] * alpha + add;
      }
    }
  }
}

// ---------------------------------------------------------------------------
// RoPE (in place, fp32): pairs (i, i+128) within each head of width 256
// ---------------------------------------------------------------------------
__global__ __launch_bounds__(256)
void rope_kernel(float* __restrict__ x, const int* __restrict__ pos,
                 int rows, int heads, int ld)
{
  const long idx = (long)blockIdx.x * blockDim.x + threadIdx.x;
  const long total = (long)rows * heads * 128;
  if (idx >= total) return;
  const int  i   = (int)(idx & 127);
  const long t   = idx >> 7;
  const int  h   = (int)(t % heads);
  const long row = t / heads;
  const float p   = (float)pos[row];
  const float inv = __powf(10000.0f, -(float)i * (1.0f / 128.0f));
  const float f   = p * inv;
  float s, c;
  __sincosf(f, &s, &c);
  float* base = x + row * (long)ld + h * 256;
  const float x1 = base[i];
  const float x2 = base[i + 128];
  base[i]       = c * x1 - s * x2;
  base[i + 128] = s * x1 + c * x2;
}

// ---------------------------------------------------------------------------
// Row softmax, in place. One 256-thread block per row of length S.
// ---------------------------------------------------------------------------
__global__ __launch_bounds__(256)
void softmax_kernel(float* __restrict__ attn, int S)
{
  float* p = attn + (long)blockIdx.x * S;
  __shared__ float red[256];
  const int tid = threadIdx.x;

  float m = -3.402823466e38f;
  for (int i = tid; i < S; i += 256) m = fmaxf(m, p[i]);
  red[tid] = m; __syncthreads();
  for (int s = 128; s > 0; s >>= 1) {
    if (tid < s) red[tid] = fmaxf(red[tid], red[tid + s]);
    __syncthreads();
  }
  m = red[0]; __syncthreads();

  float sum = 0.f;
  for (int i = tid; i < S; i += 256) {
    const float e = __expf(p[i] - m);
    p[i] = e; sum += e;
  }
  red[tid] = sum; __syncthreads();
  for (int s = 128; s > 0; s >>= 1) {
    if (tid < s) red[tid] += red[tid + s];
    __syncthreads();
  }
  const float inv = 1.0f / red[0];
  for (int i = tid; i < S; i += 256) p[i] *= inv;
}

// ---------------------------------------------------------------------------
extern "C" void kernel_launch(void* const* d_in, const int* in_sizes, int n_in,
                              void* d_out, int out_size, void* d_ws, size_t ws_size,
                              hipStream_t stream)
{
  const float* hs   = (const float*)d_in[0];
  const float* mask = (const float*)d_in[1];
  const int*   pos  = (const int*)  d_in[2];
  const float* Wq   = (const float*)d_in[3];
  const float* bq   = (const float*)d_in[4];
  const float* Wk   = (const float*)d_in[5];
  const float* bk   = (const float*)d_in[6];
  const float* Wv   = (const float*)d_in[7];
  const float* bv   = (const float*)d_in[8];
  const float* Wo   = (const float*)d_in[9];
  const float* bo   = (const float*)d_in[10];

  constexpr int Bb = 2, S = 2048, HID = 2048, H = 8, D = 256;
  constexpr int Mr = Bb * S;                       // 4096 rows

  float* out  = (float*)d_out;                     // (B,S,HID)
  float* attn = out + (long)Bb * S * HID;          // (B,H,S,S)

  float* q  = (float*)d_ws;                        // (Mr, H*D)
  float* k  = q  + (long)Mr * H * D;               // (Mr, D)
  float* v  = k  + (long)Mr * D;                   // (Mr, D)
  float* ao = v  + (long)Mr * D;                   // (Mr, H*D) attn-out

  const dim3 blk(256);

  // 1) QKV projections: C = hs @ W^T + b
  gemm_wmma<true><<<dim3((H*D)/128, Mr/128, 1), blk, 0, stream>>>(
      hs, HID, 0, 0,  Wq, HID, 0, 0,  q, H*D, 0, 0,
      Mr, H*D, HID,  bq, 1, 0,  1.0f, 1);
  gemm_wmma<true><<<dim3(D/128, Mr/128, 1), blk, 0, stream>>>(
      hs, HID, 0, 0,  Wk, HID, 0, 0,  k, D, 0, 0,
      Mr, D, HID,  bk, 1, 0,  1.0f, 1);
  gemm_wmma<true><<<dim3(D/128, Mr/128, 1), blk, 0, stream>>>(
      hs, HID, 0, 0,  Wv, HID, 0, 0,  v, D, 0, 0,
      Mr, D, HID,  bv, 1, 0,  1.0f, 1);

  // 2) RoPE on q (H heads) and k (1 head)
  rope_kernel<<<(Mr * H * 128) / 256, blk, 0, stream>>>(q, pos, Mr, H, H * D);
  rope_kernel<<<(Mr * 1 * 128) / 256, blk, 0, stream>>>(k, pos, Mr, 1, D);

  // 3) Scores (batched over 16 (b,h)): attn = (q . k^T)/16 + mask[b, key]
  gemm_wmma<true><<<dim3(S/128, S/128, Bb*H), blk, 0, stream>>>(
      q,    H*D, (long)S * H * D, (long)D,
      k,    D,   (long)S * D,     0,
      attn, S,   (long)H * S * S, (long)S * S,
      S, S, D,
      mask, H, (long)S,
      0.0625f /* D^-0.5 */, H);

  // 4) Softmax rows in place on d_out attn region
  softmax_kernel<<<Bb * H * S, blk, 0, stream>>>(attn, S);

  // 5) attn @ V (batched, B operand stored [k][n]) -> head-interleaved ao
  gemm_wmma<false><<<dim3(D/128, S/128, Bb*H), blk, 0, stream>>>(
      attn, S,   (long)H * S * S, (long)S * S,
      v,    D,   (long)S * D,     0,
      ao,   H*D, (long)S * H * D, (long)D,
      S, D, S,
      nullptr, 1, 0,
      1.0f, H);

  // 6) Output projection: out = ao @ Wo^T + bo
  gemm_wmma<true><<<dim3(HID/128, Mr/128, 1), blk, 0, stream>>>(
      ao, H*D, 0, 0,  Wo, H*D, 0, 0,  out, HID, 0, 0,
      Mr, HID, H*D,  bo, 1, 0,  1.0f, 1);
}
